// PNA_4166118277517
// MI455X (gfx1250) — matrix operand order repaired and checked by reading, not compiled, not run on hardware
//
#include <hip/hip_runtime.h>

// ---------------------------------------------------------------------------
// PNA forward for MI455X (gfx1250). wave32, WMMA f32<-f16 16x16x32.
// B matrices are pre-packed into per-lane fragment layout so the GEMM inner
// loop is pure b128 loads + v_wmma.
// ---------------------------------------------------------------------------

typedef __attribute__((ext_vector_type(16))) _Float16 v16h;
typedef __attribute__((ext_vector_type(8)))  _Float16 h8;
typedef __attribute__((ext_vector_type(8)))  float    v8f;

#define DEV __device__ __forceinline__

static constexpr int   Nn   = 20000;
static constexpr int   Ee   = 320000;
static constexpr int   CH   = 128;   // hidden channels
static constexpr int   TOW  = 4;     // towers (hidden layers)
static constexpr int   FIH  = 32;    // per-tower width (hidden)
static constexpr int   CLS  = 40;    // classes
static constexpr int   CLSP = 64;    // classes padded to multiple of 32 (K-safe)
static constexpr int   LAYS = 3;     // hidden conv layers
static constexpr int   FEATW = 13 * CH; // 1664 f16 per node feature row

// ---------------- WMMA helpers (CDNA5 layouts) -----------------------------

DEV v8f wmma_fp16(v16h a, v16h b, v8f c) {
  return __builtin_amdgcn_wmma_f32_16x16x32_f16(false, a, false, b, (short)0, c,
                                                false, false);
}

// A fragment: 16x32 f16, row pointer per lane (gather-friendly).
// lanes 0-15 hold K 0..7 / 16..23, lanes 16-31 hold K 8..15 / 24..31.
DEV v16h a_frag(const _Float16* __restrict__ rowptr, int lane) {
  int kb = (lane >> 4) << 3;                   // 0 or 8
  h8 lo = *(const h8*)(rowptr + kb);           // K = kb .. kb+7
  h8 hi = *(const h8*)(rowptr + 16 + kb);      // K = 16+kb .. 16+kb+7
  v16h a;
#pragma unroll
  for (int j = 0; j < 8; ++j) { a[j] = lo[j]; a[8 + j] = hi[j]; }
  return a;
}

// Packed-B fragment: pk holds, for each 32x16 tile, 32 lanes x 16 halves
// contiguously. Two aligned b128 loads per fragment.
DEV v16h b_frag_pk(const _Float16* __restrict__ pk, int kt, int ct, int ntiles,
                   int lane) {
  const _Float16* p = pk + ((((size_t)kt * ntiles + ct) * 32 + lane) << 4);
  h8 lo = *(const h8*)p;
  h8 hi = *(const h8*)(p + 8);
  v16h b;
#pragma unroll
  for (int j = 0; j < 8; ++j) { b[j] = lo[j]; b[8 + j] = hi[j]; }
  return b;
}

// Order-preserving float<->uint key for atomic min/max on floats.
DEV unsigned fkey(float v) {
  unsigned u = __float_as_uint(v);
  return (u & 0x80000000u) ? ~u : (u | 0x80000000u);
}
DEV float fdecode(unsigned k) {
  return __uint_as_float((k & 0x80000000u) ? (k ^ 0x80000000u) : ~k);
}

// ---------------- utility kernels ------------------------------------------

__global__ void k_cvt_pad(const float* __restrict__ src, _Float16* __restrict__ dst,
                          long srows, int scols, long drows, int dcols) {
  long i = (long)blockIdx.x * blockDim.x + threadIdx.x;
  long tot = drows * (long)dcols;
  if (i >= tot) return;
  long r = i / dcols; int c = (int)(i % dcols);
  float v = (r < srows && c < scols) ? src[r * (long)scols + c] : 0.f;
  dst[i] = (_Float16)v;
}

// Pack B [Kv,Nv] f32 (row-major) -> fragment layout f16, zero-padded to Kp,Np.
// dst[b][ (kt*ntiles+ct)*32 + lane ][ j ] = B[kt*32 + (lane>>4)*8 + (j&7) + (j>>3)*16][ct*16 + lane&15]
__global__ void k_pack_b(const float* __restrict__ src, _Float16* __restrict__ dst,
                         int Kv, int Nv, int Kp, int Np, long srcBatch, long total) {
  long i = (long)blockIdx.x * blockDim.x + threadIdx.x;
  if (i >= total) return;
  long per = (long)Kp * Np;
  long b = i / per;
  long rem = i - b * per;
  int j    = (int)(rem & 15);
  int lane = (int)((rem >> 4) & 31);
  int tile = (int)(rem >> 9);
  int ntiles = Np >> 4;
  int kt = tile / ntiles, ct = tile - kt * ntiles;
  int k = kt * 32 + ((lane >> 4) << 3) + (j & 7) + ((j >> 3) << 4);
  int n = ct * 16 + (lane & 15);
  float v = (k < Kv && n < Nv) ? src[b * srcBatch + (size_t)k * Nv + n] : 0.f;
  dst[i] = (_Float16)v;
}

__global__ void k_fill_f32(float* __restrict__ p, float v, long n) {
  long i = (long)blockIdx.x * blockDim.x + threadIdx.x;
  if (i < n) p[i] = v;
}
__global__ void k_fill_u32(unsigned* __restrict__ p, unsigned v, long n) {
  long i = (long)blockIdx.x * blockDim.x + threadIdx.x;
  if (i < n) p[i] = v;
}

__global__ void k_degree(const int* __restrict__ dstI, float* __restrict__ cnt) {
  int e = blockIdx.x * blockDim.x + threadIdx.x;
  if (e < Ee) atomicAdd(&cnt[dstI[e]], 1.f);
}

__global__ void k_avglog(const float* __restrict__ cnt, float* __restrict__ acc) {
  int i = blockIdx.x * blockDim.x + threadIdx.x;
  if (i < Nn) atomicAdd(acc, logf(cnt[i] + 1.f));
}

// ---------------- generic WMMA GEMM ----------------------------------------
// One wave per 16x16 output tile. blockIdx = (Mtile, Ntile, batch z).
// B is pre-packed; bNtiles = Np/16 of the packed matrix.
__global__ void k_gemm16(const _Float16* __restrict__ A, int lda, int aBatch,
                         const _Float16* __restrict__ Bpk, int bNtiles, long bBatch,
                         int K,
                         float* __restrict__ oF32, _Float16* __restrict__ oF16,
                         int ldc, int colBatch,
                         const float* __restrict__ bias, int biasBatch, int validCols,
                         int epilogue,
                         const float* __restrict__ g, const float* __restrict__ bt,
                         const float* __restrict__ rm, const float* __restrict__ rv) {
  const int lane  = threadIdx.x & 31;
  const int mtile = blockIdx.x;
  const int ctile = blockIdx.y;
  const int z     = blockIdx.z;

  const _Float16* Ab = A + (size_t)aBatch * z;
  const _Float16* Bb = Bpk + (size_t)bBatch * z;
  const int mrow = mtile * 16 + (lane & 15);
  const _Float16* arow = Ab + (size_t)mrow * lda;

  v8f acc = {};
  for (int k = 0; k < K; k += 32) {
    v16h a = a_frag(arow + k, lane);
    v16h b = b_frag_pk(Bb, k >> 5, ctile, bNtiles, lane);
    acc = wmma_fp16(a, b, acc);
  }

  const int n      = lane & 15;
  const int colInB = ctile * 16 + n;
  const bool valid = colInB < validCols;
  const int gcol   = z * colBatch + colInB;
  float bv = valid ? bias[(size_t)biasBatch * z + colInB] : 0.f;
  float bnS = 1.f, bnO = 0.f;
  if (epilogue == 1) {
    float inv = rsqrtf(rv[gcol] + 1e-5f);
    bnS = inv * g[gcol];
    bnO = bt[gcol] - rm[gcol] * bnS;
  }
  const int rbase = mtile * 16 + (lane >> 4) * 8;
#pragma unroll
  for (int r = 0; r < 8; ++r) {
    float v = acc[r] + bv;
    if (epilogue == 1) v = fmaxf(v * bnS + bnO, 0.f);
    if (!valid) v = 0.f;
    size_t o = (size_t)(rbase + r) * ldc + gcol;
    if (oF32) oF32[o] = v;
    if (oF16) oF16[o] = (_Float16)v;
  }
}

// ---------------- fused edge pre-MLP + segment aggregation -----------------
// m[e,t,:] = [x_t[dst]|x_t[src]|e_enc] @ Wpre_t + bpre_t via WMMA (16 edges
// per block, 8 waves = NT towers x FI/16 column tiles), then sum/sum^2/min/max
// go straight into node accumulators with atomics -- no [E,128] buffer in HBM.
template <int NT, int FI>
__global__ void edge_pre_agg(const _Float16* __restrict__ xh,
                             const _Float16* __restrict__ eenc,
                             const int* __restrict__ srcI, const int* __restrict__ dstI,
                             const _Float16* __restrict__ WprePk, // packed, per-tower
                             const float* __restrict__ bpre,      // [NT*FI]
                             float* __restrict__ sumB, float* __restrict__ sumsqB,
                             unsigned* __restrict__ minK, unsigned* __restrict__ maxK) {
  constexpr int C  = NT * FI;   // 128
  constexpr int CT = FI / 16;   // col tiles per tower
  const int lane = threadIdx.x & 31;
  const int wave = threadIdx.x >> 5;
  const int t  = wave / CT;
  const int wc = wave % CT;
  const int etile = blockIdx.x;
  const int e = etile * 16 + (lane & 15);
  const int dnode = dstI[e];
  const int snode = srcI[e];

  const _Float16* rb0 = xh + (size_t)dnode * C + t * FI;  // xi
  const _Float16* rb1 = xh + (size_t)snode * C + t * FI;  // xj
  const _Float16* rb2 = eenc + (size_t)e * FI;            // e (tower-broadcast)
  const _Float16* rbs[3] = {rb0, rb1, rb2};
  const _Float16* Wt = WprePk + (size_t)t * (3 * FI * FI);  // packed tower stride

  v8f acc = {};
#pragma unroll
  for (int s = 0; s < 3; ++s) {
#pragma unroll
    for (int ks = 0; ks < FI / 32; ++ks) {
      v16h a = a_frag(rbs[s] + ks * 32, lane);
      v16h b = b_frag_pk(Wt, s * (FI / 32) + ks, wc, CT, lane);
      acc = wmma_fp16(a, b, acc);
    }
  }

  const int colF = wc * 16 + (lane & 15);
  const int gcol = t * FI + colF;
  const float bv = bpre[t * FI + colF];
  const int rbase = etile * 16 + (lane >> 4) * 8;
#pragma unroll
  for (int r = 0; r < 8; ++r) {
    int er = rbase + r;
    int dn = dstI[er];
    float v = acc[r] + bv;
    size_t idx = (size_t)dn * C + gcol;
    atomicAdd(&sumB[idx], v);
    atomicAdd(&sumsqB[idx], v * v);
    unsigned key = fkey(v);
    atomicMax(&maxK[idx], key);
    atomicMin(&minK[idx], key);
  }
}

// ---------------- PNA scaler / feature builder -----------------------------
__global__ void k_feat(const _Float16* __restrict__ xh,
                       const float* __restrict__ sumB, const float* __restrict__ sumsqB,
                       const unsigned* __restrict__ minK, const unsigned* __restrict__ maxK,
                       const float* __restrict__ cnt, const float* __restrict__ avgAcc,
                       _Float16* __restrict__ feat, int FI) {
  long i = (long)blockIdx.x * blockDim.x + threadIdx.x;
  if (i >= (long)Nn * CH) return;
  int node = (int)(i >> 7);
  int c = (int)(i & 127);
  int t = c / FI, kk = c % FI;

  float cn = cnt[node];
  float c1 = fmaxf(cn, 1.f);
  float mean = sumB[i] / c1;
  float var  = sumsqB[i] / c1 - mean * mean;
  float sd   = sqrtf(fmaxf(var, 0.f) + 1e-5f);
  bool  has  = cn > 0.f;
  float mn = has ? fdecode(minK[i]) : 0.f;
  float mx = has ? fdecode(maxK[i]) : 0.f;
  float ld  = logf(c1 + 1.f);
  float avg = avgAcc[0] / (float)Nn;
  float amp = ld / avg, att = avg / ld;

  _Float16* row = feat + (size_t)node * FEATW + (size_t)t * (13 * FI);
  row[kk] = xh[(size_t)node * CH + c];
  float st[4] = {mean, mn, mx, sd};
#pragma unroll
  for (int q = 0; q < 4; ++q) {
    row[(1 + q) * FI + kk] = (_Float16)st[q];
    row[(5 + q) * FI + kk] = (_Float16)(st[q] * amp);
    row[(9 + q) * FI + kk] = (_Float16)(st[q] * att);
  }
}

// ---------------- log-softmax ----------------------------------------------
__global__ void k_logsoftmax(const float* __restrict__ logits, float* __restrict__ out) {
  int node = blockIdx.x * blockDim.x + threadIdx.x;
  if (node >= Nn) return;
  const float* r = logits + (size_t)node * CLSP;
  float mx = -3.4e38f;
  for (int j = 0; j < CLS; ++j) mx = fmaxf(mx, r[j]);
  float s = 0.f;
  for (int j = 0; j < CLS; ++j) s += expf(r[j] - mx);
  float lse = mx + logf(s);
  float* o = out + (size_t)node * CLS;
  for (int j = 0; j < CLS; ++j) o[j] = r[j] - lse;
}

// ---------------------------------------------------------------------------

static inline unsigned nblk(long n, int b) { return (unsigned)((n + b - 1) / b); }

extern "C" void kernel_launch(void* const* d_in, const int* in_sizes, int n_in,
                              void* d_out, int out_size, void* d_ws, size_t ws_size,
                              hipStream_t stream) {
  (void)in_sizes; (void)n_in; (void)out_size; (void)ws_size;
  const float* x       = (const float*)d_in[0];
  const int*   eidx    = (const int*)d_in[1];
  const float* eattr   = (const float*)d_in[2];
  const float* We_h    = (const float*)d_in[3];
  const float* be_h    = (const float*)d_in[4];
  const float* Wpre_h  = (const float*)d_in[5];
  const float* bpre_h  = (const float*)d_in[6];
  const float* Wpost_h = (const float*)d_in[7];
  const float* bpost_h = (const float*)d_in[8];
  const float* Wlin_h  = (const float*)d_in[9];
  const float* blin_h  = (const float*)d_in[10];
  const float* gamma   = (const float*)d_in[11];
  const float* beta    = (const float*)d_in[12];
  const float* rmean   = (const float*)d_in[13];
  const float* rvar    = (const float*)d_in[14];
  const float* We_f    = (const float*)d_in[15];
  const float* be_f    = (const float*)d_in[16];
  const float* Wpre_f  = (const float*)d_in[17];
  const float* bpre_f  = (const float*)d_in[18];
  const float* Wpost_f = (const float*)d_in[19];
  const float* bpost_f = (const float*)d_in[20];
  const float* Wlin_f  = (const float*)d_in[21];
  const float* blin_f  = (const float*)d_in[22];
  const int* srcI = eidx;
  const int* dstI = eidx + Ee;

  // --- workspace carve-out -------------------------------------------------
  char* ws = (char*)d_ws;
  size_t off = 0;
  auto take = [&](size_t bytes) -> char* {
    char* p = ws + off;
    off = (off + bytes + 255) & ~(size_t)255;
    return p;
  };
  _Float16* xh     = (_Float16*)take((size_t)Nn * CH * 2);        // node feats f16
  _Float16* ea16   = (_Float16*)take((size_t)Ee * 32 * 2);        // edge_attr padded
  _Float16* eenc16 = (_Float16*)take((size_t)Ee * CH * 2);        // edge encoder out
  _Float16* t1h    = (_Float16*)take((size_t)Nn * CH * 2);        // post-MLP out f16
  _Float16* feat   = (_Float16*)take((size_t)Nn * FEATW * 2);     // PNA features
  float*    logits = (float*)take((size_t)Nn * CLSP * 4);
  float*    cnt    = (float*)take((size_t)Nn * 4);
  float*    avgAcc = (float*)take(256);
  float*    sumB   = (float*)take((size_t)Nn * CH * 4);
  float*    sumsqB = (float*)take((size_t)Nn * CH * 4);
  unsigned* minK   = (unsigned*)take((size_t)Nn * CH * 4);
  unsigned* maxK   = (unsigned*)take((size_t)Nn * CH * 4);
  _Float16* wWe    = (_Float16*)take((size_t)32 * CH * 2);           // packed
  _Float16* wPre   = (_Float16*)take((size_t)3 * CH * CH * 2);       // packed
  _Float16* wPost  = (_Float16*)take((size_t)13 * CH * CLSP * 2);    // packed
  _Float16* wLin   = (_Float16*)take((size_t)CH * CH * 2);           // packed

  const dim3 B256(256);
  const dim3 W32(32);
  const float* nofp = nullptr;

  // --- one-time prep -------------------------------------------------------
  k_cvt_pad<<<nblk((long)Nn * CH, 256), B256, 0, stream>>>(x, xh, Nn, CH, Nn, CH);
  k_cvt_pad<<<nblk((long)Ee * 32, 256), B256, 0, stream>>>(eattr, ea16, Ee, 16, Ee, 32);
  k_fill_f32<<<nblk(Nn, 256), B256, 0, stream>>>(cnt, 0.f, Nn);
  k_degree<<<nblk(Ee, 256), B256, 0, stream>>>(dstI, cnt);
  k_fill_f32<<<1, B256, 0, stream>>>(avgAcc, 0.f, 1);
  k_avglog<<<nblk(Nn, 256), B256, 0, stream>>>(cnt, avgAcc);

  const long sC = (long)Nn * CH;

  // --- hidden layers -------------------------------------------------------
  for (int i = 0; i < LAYS; ++i) {
    // pack weights (f32 -> f16 fragment layout, zero-padded)
    long tWe = 32 * FIH;                         // Kp=32, Np=32
    k_pack_b<<<nblk(tWe, 256), B256, 0, stream>>>(
        We_h + (size_t)i * 16 * FIH, wWe, 16, FIH, 32, FIH, 0, tWe);
    long tPre = (long)TOW * 3 * FIH * FIH;       // per tower Kp=96, Np=32
    k_pack_b<<<nblk(tPre, 256), B256, 0, stream>>>(
        Wpre_h + (size_t)i * TOW * 3 * FIH * FIH, wPre,
        3 * FIH, FIH, 3 * FIH, FIH, (long)3 * FIH * FIH, tPre);
    long tPost = (long)TOW * 13 * FIH * FIH;     // per tower Kp=416, Np=32
    k_pack_b<<<nblk(tPost, 256), B256, 0, stream>>>(
        Wpost_h + (size_t)i * TOW * 13 * FIH * FIH, wPost,
        13 * FIH, FIH, 13 * FIH, FIH, (long)13 * FIH * FIH, tPost);
    long tLin = (long)CH * CH;
    k_pack_b<<<nblk(tLin, 256), B256, 0, stream>>>(
        Wlin_h + (size_t)i * CH * CH, wLin, CH, CH, CH, CH, 0, tLin);

    // edge encoder: e_enc = edge_attr @ We + be  -> [E,32] f16
    k_gemm16<<<dim3(Ee / 16, FIH / 16, 1), W32, 0, stream>>>(
        ea16, 32, 0, wWe, FIH / 16, 0, 32,
        nullptr, eenc16, FIH, 0,
        be_h + (size_t)i * FIH, 0, 1 << 30, 0, nofp, nofp, nofp, nofp);

    // reset aggregation state
    k_fill_f32<<<nblk(sC, 256), B256, 0, stream>>>(sumB, 0.f, sC);
    k_fill_f32<<<nblk(sC, 256), B256, 0, stream>>>(sumsqB, 0.f, sC);
    k_fill_u32<<<nblk(sC, 256), B256, 0, stream>>>(minK, 0xFFFFFFFFu, sC);
    k_fill_u32<<<nblk(sC, 256), B256, 0, stream>>>(maxK, 0u, sC);

    // fused edge pre-MLP (WMMA) + atomic segment aggregation
    edge_pre_agg<TOW, FIH><<<dim3(Ee / 16), B256, 0, stream>>>(
        xh, eenc16, srcI, dstI, wPre, bpre_h + (size_t)i * CH,
        sumB, sumsqB, minK, maxK);

    // scalers + feature rows
    k_feat<<<nblk(sC, 256), B256, 0, stream>>>(
        xh, sumB, sumsqB, minK, maxK, cnt, avgAcc, feat, FIH);

    // per-tower post-MLP: [N,416] @ [416,32], batched z = tower
    k_gemm16<<<dim3(Nn / 16, FIH / 16, TOW), W32, 0, stream>>>(
        feat, FEATW, 13 * FIH, wPost, FIH / 16, (long)13 * FIH * FIH, 13 * FIH,
        nullptr, t1h, CH, FIH,
        bpost_h + (size_t)i * CH, FIH, 1 << 30, 0, nofp, nofp, nofp, nofp);

    // Wlin + fused BatchNorm + ReLU -> next-layer xh (f16)
    k_gemm16<<<dim3(Nn / 16, CH / 16, 1), W32, 0, stream>>>(
        t1h, CH, 0, wLin, CH / 16, 0, CH,
        nullptr, xh, CH, 0,
        blin_h + (size_t)i * CH, 0, 1 << 30, 1,
        gamma + (size_t)i * CH, beta + (size_t)i * CH,
        rmean + (size_t)i * CH, rvar + (size_t)i * CH);
  }

  // --- final PNAConv (towers=1, fi=128, out=40 padded to 64) ---------------
  long tWeF = 32 * CH;
  k_pack_b<<<nblk(tWeF, 256), B256, 0, stream>>>(We_f, wWe, 16, CH, 32, CH, 0, tWeF);
  long tPreF = (long)3 * CH * CH;
  k_pack_b<<<nblk(tPreF, 256), B256, 0, stream>>>(
      Wpre_f, wPre, 3 * CH, CH, 3 * CH, CH, 0, tPreF);
  long tPostF = (long)13 * CH * CLSP;
  k_pack_b<<<nblk(tPostF, 256), B256, 0, stream>>>(
      Wpost_f, wPost, 13 * CH, CLS, 13 * CH, CLSP, 0, tPostF);
  long tLinF = (long)CLSP * CLSP;
  k_pack_b<<<nblk(tLinF, 256), B256, 0, stream>>>(
      Wlin_f, wLin, CLS, CLS, CLSP, CLSP, 0, tLinF);

  // edge encoder: [E,16(->32)] @ [32,128] -> [E,128] f16
  k_gemm16<<<dim3(Ee / 16, CH / 16, 1), W32, 0, stream>>>(
      ea16, 32, 0, wWe, CH / 16, 0, 32,
      nullptr, eenc16, CH, 0,
      be_f, 0, 1 << 30, 0, nofp, nofp, nofp, nofp);

  k_fill_f32<<<nblk(sC, 256), B256, 0, stream>>>(sumB, 0.f, sC);
  k_fill_f32<<<nblk(sC, 256), B256, 0, stream>>>(sumsqB, 0.f, sC);
  k_fill_u32<<<nblk(sC, 256), B256, 0, stream>>>(minK, 0xFFFFFFFFu, sC);
  k_fill_u32<<<nblk(sC, 256), B256, 0, stream>>>(maxK, 0u, sC);

  edge_pre_agg<1, CH><<<dim3(Ee / 16), B256, 0, stream>>>(
      xh, eenc16, srcI, dstI, wPre, bpre_f, sumB, sumsqB, minK, maxK);

  k_feat<<<nblk(sC, 256), B256, 0, stream>>>(
      xh, sumB, sumsqB, minK, maxK, cnt, avgAcc, feat, CH);

  // post-MLP: [N,1664] @ [1664,64(valid 40)] -> t1h [N,64] f16
  k_gemm16<<<dim3(Nn / 16, CLSP / 16, 1), W32, 0, stream>>>(
      feat, FEATW, 0, wPost, CLSP / 16, 0, 13 * CH,
      nullptr, t1h, CLSP, 0,
      bpost_f, 0, CLS, 0, nofp, nofp, nofp, nofp);

  // final linear: [N,64] @ [64,64(valid 40)] -> logits f32
  k_gemm16<<<dim3(Nn / 16, CLSP / 16, 1), W32, 0, stream>>>(
      t1h, CLSP, 0, wLin, CLSP / 16, 0, CLSP,
      logits, nullptr, CLSP, 0,
      blin_f, 0, CLS, 0, nofp, nofp, nofp, nofp);

  k_logsoftmax<<<nblk(Nn, 256), B256, 0, stream>>>(logits, (float*)d_out);
}
